// QLinear_28415503631119
// MI455X (gfx1250) — matrix operand order repaired and checked by reading, not compiled, not run on hardware
//
#include <hip/hip_runtime.h>
#include <stdint.h>

// ---------------------------------------------------------------------------
// Problem constants (match reference)
// ---------------------------------------------------------------------------
static constexpr int Bt  = 4096;   // tokens (M)
static constexpr int Id  = 4096;   // in features (K)
static constexpr int Od  = 4096;   // out features (N)
static constexpr int GS  = 16;     // quant group size
static constexpr int Gd  = Id / GS;      // 256 groups per row
static constexpr int NNZ = 32768;

// GEMM tiling
static constexpr int BM  = 128;
static constexpr int BN  = 128;
static constexpr int BK  = 64;
static constexpr int BKP = BK + 8;       // padded LDS row (144 B, 16B-aligned)
static constexpr int KCH = Id / BK;      // 64 K-chunks

// ---------------------------------------------------------------------------
// Types for WMMA
// ---------------------------------------------------------------------------
typedef __attribute__((ext_vector_type(8)))  __bf16 v8bf;
typedef __attribute__((ext_vector_type(16))) __bf16 v16bf;
typedef __attribute__((ext_vector_type(8)))  float  v8f;

union Frag16 {
  v16bf v16;
  v8bf  v8[2];
};

// ---------------------------------------------------------------------------
// bf16 helpers (round-to-nearest-even)
// ---------------------------------------------------------------------------
__device__ __forceinline__ unsigned short f32_to_bf16(float f) {
  unsigned int u = __float_as_uint(f);
  u += 0x7FFFu + ((u >> 16) & 1u);
  return (unsigned short)(u >> 16);
}
__device__ __forceinline__ float bf16_to_f32(unsigned short b) {
  return __uint_as_float(((unsigned int)b) << 16);
}

// ---------------------------------------------------------------------------
// CDNA5 async global->LDS (per-lane 16B), tracked by ASYNCcnt.
//   LDS offset taken from a VGPR; global address from a 64-bit VGPR pair
// ---------------------------------------------------------------------------
__device__ __forceinline__ void async_load_b128(unsigned lds_off, const void* gptr) {
  asm volatile("global_load_async_to_lds_b128 %0, %1, off"
               :
               : "v"(lds_off), "v"(gptr)
               : "memory");
}
__device__ __forceinline__ void wait_asynccnt0() {
  asm volatile("s_wait_asynccnt 0x0" ::: "memory");
}

// ---------------------------------------------------------------------------
// Kernel 1: x_bf[b, i] = bf16(input[b, perm[i]])
// ---------------------------------------------------------------------------
__global__ __launch_bounds__(256) void k_permute_x(
    const float* __restrict__ input, const int* __restrict__ perm,
    unsigned short* __restrict__ xbf) {
  size_t idx = (size_t)blockIdx.x * 256 + threadIdx.x;   // < Bt*Id
  size_t b = idx >> 12;          // / 4096
  int    i = (int)(idx & 4095);
  xbf[idx] = f32_to_bf16(input[(b << 12) + (size_t)perm[i]]);
}

// ---------------------------------------------------------------------------
// Kernel 2: dequantize W -> bf16.
//   col j        <- hi nibble of qweight[o][j]
//   col j + I/2  <- lo nibble of qweight[o][j]
// W[o,i] = scale[o,g] * (qw - zero[o,g]),  g = i/16
// ---------------------------------------------------------------------------
__global__ __launch_bounds__(256) void k_dequant_w(
    const int* __restrict__ qweight,
    const int* __restrict__ scale_q, const float* __restrict__ sqs, const float* __restrict__ sqz,
    const int* __restrict__ zero_q,  const float* __restrict__ zqs, const float* __restrict__ zqz,
    unsigned short* __restrict__ wbf) {
  size_t t = (size_t)blockIdx.x * 256 + threadIdx.x;     // < Od * Id/2
  size_t o = t >> 11;            // / 2048
  int    j = (int)(t & 2047);

  int q    = qweight[t];
  float qh = (float)((q >> 4) & 15);
  float ql = (float)(q & 15);

  size_t og1 = o * (size_t)Gd + (size_t)(j >> 4);        // group of col j
  size_t og2 = og1 + (size_t)((Id / 2) / GS);            // group of col j + I/2

  float s1 = sqs[og1] * ((float)scale_q[og1] - sqz[og1]);
  float z1 = zqs[og1] * ((float)zero_q[og1]  - zqz[og1]);
  float s2 = sqs[og2] * ((float)scale_q[og2] - sqz[og2]);
  float z2 = zqs[og2] * ((float)zero_q[og2]  - zqz[og2]);

  size_t base = (o << 12) + (size_t)j;
  wbf[base]            = f32_to_bf16(s1 * (qh - z1));
  wbf[base + (Id / 2)] = f32_to_bf16(s2 * (ql - z2));
}

// ---------------------------------------------------------------------------
// Kernel 3: zero outlier positions (matches reference .set(0.0)).
// ---------------------------------------------------------------------------
__global__ __launch_bounds__(256) void k_zero_outliers(
    unsigned short* __restrict__ wbf,
    const int* __restrict__ rows, const int* __restrict__ cols) {
  int k = blockIdx.x * 256 + threadIdx.x;
  if (k >= NNZ) return;
  wbf[((size_t)rows[k] << 12) + (size_t)cols[k]] = 0;
}

// ---------------------------------------------------------------------------
// Kernel 4: fold sparse outliers into W: W[r,c] += val (CAS bf16 add;
// duplicates sum exactly like segment_sum on top of the zeroed slot).
// ---------------------------------------------------------------------------
__global__ __launch_bounds__(256) void k_add_outliers(
    unsigned short* __restrict__ wbf,
    const int* __restrict__ rows, const int* __restrict__ cols,
    const float* __restrict__ vals) {
  int k = blockIdx.x * 256 + threadIdx.x;
  if (k >= NNZ) return;
  size_t idx = ((size_t)rows[k] << 12) + (size_t)cols[k];
  unsigned short* hp = wbf + idx;
  unsigned int* word = (unsigned int*)((size_t)hp & ~(size_t)3);
  bool hi = (((size_t)hp) & 2) != 0;
  float v = vals[k];

  unsigned int old = *word, assumed;
  do {
    assumed = old;
    unsigned short cb = hi ? (unsigned short)(assumed >> 16)
                           : (unsigned short)(assumed & 0xFFFFu);
    unsigned short nb = f32_to_bf16(bf16_to_f32(cb) + v);
    unsigned int nw = hi ? ((assumed & 0x0000FFFFu) | ((unsigned int)nb << 16))
                         : ((assumed & 0xFFFF0000u) | (unsigned int)nb);
    old = atomicCAS(word, assumed, nw);
  } while (old != assumed);
}

// ---------------------------------------------------------------------------
// Kernel 5: GEMM  out[b,o] = sum_i x_bf[b,i] * w_bf[o,i] + bias[o]
// 256 threads = 8 wave32s. Block tile 128x128, K-chunk 64.
// Wave grid 2(M) x 4(N): each wave owns 64x32 = 4x2 WMMA 16x16 tiles.
// Double-buffered LDS filled by GLOBAL_LOAD_ASYNC_TO_LDS_B128 (zero VGPR
// staging -> no spills), fenced with s_wait_asynccnt + barrier.
// ---------------------------------------------------------------------------
__global__ __launch_bounds__(256) void k_gemm_bf16(
    const unsigned short* __restrict__ xbf,
    const unsigned short* __restrict__ wbf,
    const float* __restrict__ bias,
    float* __restrict__ out) {
  __shared__ __align__(16) unsigned short Xs[2][BM][BKP];
  __shared__ __align__(16) unsigned short Ws[2][BN][BKP];

  const int tid = threadIdx.x;
  const int on0 = blockIdx.x * BN;     // output-feature tile
  const int bm0 = blockIdx.y * BM;     // token tile

  // cooperative-load mapping: 8 lanes x 16B per row, 32 rows per pass, 4 passes
  const int sub  = tid & 7;            // which 16B chunk of the 128B row
  const int rowb = tid >> 3;           // 0..31

  // wave/lane mapping for WMMA
  const int w    = tid >> 5;           // 0..7
  const int lane = tid & 31;
  const int wm   = w >> 2;             // 0..1  (M)
  const int wn   = w & 3;              // 0..3  (N)
  const int nl   = lane & 15;
  const int h    = lane >> 4;

  // per-thread global source pointers (advance by BK halfs per chunk)
  const unsigned short* gx[4];
  const unsigned short* gw[4];
#pragma unroll
  for (int p = 0; p < 4; ++p) {
    int r = rowb + 32 * p;
    gx[p] = xbf + ((size_t)(bm0 + r) << 12) + sub * 8;
    gw[p] = wbf + ((size_t)(on0 + r) << 12) + sub * 8;
  }
  // per-thread LDS destination offsets (wave-relative low 32 bits)
  const unsigned lx0 = (unsigned)(uintptr_t)(&Xs[0][rowb][sub * 8]);
  const unsigned lw0 = (unsigned)(uintptr_t)(&Ws[0][rowb][sub * 8]);
  const unsigned bufStride = (unsigned)(BM * BKP * sizeof(unsigned short));
  const unsigned rowStride = (unsigned)(32 * BKP * sizeof(unsigned short));

  auto issue_async = [&](int buf) {
    unsigned bx = lx0 + (unsigned)buf * bufStride;
    unsigned bw = lw0 + (unsigned)buf * bufStride;
#pragma unroll
    for (int p = 0; p < 4; ++p) {
      async_load_b128(bx + (unsigned)p * rowStride, gx[p]);
      async_load_b128(bw + (unsigned)p * rowStride, gw[p]);
    }
#pragma unroll
    for (int p = 0; p < 4; ++p) { gx[p] += BK; gw[p] += BK; }
  };

  v8f acc[4][2];
  {
    v8f zero = {};
#pragma unroll
    for (int mt = 0; mt < 4; ++mt)
#pragma unroll
      for (int nt = 0; nt < 2; ++nt) acc[mt][nt] = zero;
  }

  auto compute = [&](int buf) {
#pragma unroll
    for (int ks = 0; ks < BK; ks += 32) {
      // A fragments: lane L holds row (L&15); K = {8h..8h+7} U {16+8h..+7}
      Frag16 a[4];
#pragma unroll
      for (int mt = 0; mt < 4; ++mt) {
        const unsigned short* p = &Xs[buf][wm * 64 + mt * 16 + nl][ks + 8 * h];
        a[mt].v8[0] = *(const v8bf*)(p);
        a[mt].v8[1] = *(const v8bf*)(p + 16);
      }
      // B fragments: lane L holds col (L&15); K = {8h..8h+7} U {16+8h..+7}
      Frag16 bfr[2];
#pragma unroll
      for (int nt = 0; nt < 2; ++nt) {
        const unsigned short* p = &Ws[buf][wn * 32 + nt * 16 + nl][ks + 8 * h];
        bfr[nt].v8[0] = *(const v8bf*)(p);
        bfr[nt].v8[1] = *(const v8bf*)(p + 16);
      }
#pragma unroll
      for (int mt = 0; mt < 4; ++mt)
#pragma unroll
        for (int nt = 0; nt < 2; ++nt)
          acc[mt][nt] = __builtin_amdgcn_wmma_f32_16x16x32_bf16(
              false, a[mt].v16, false, bfr[nt].v16,
              (short)0, acc[mt][nt], false, false);
    }
  };

  // prologue: fill buffer 0
  issue_async(0);
  wait_asynccnt0();
  __syncthreads();

  int cur = 0;
  for (int kc = 0; kc < KCH; ++kc) {
    if (kc + 1 < KCH) issue_async(cur ^ 1);    // fetch next chunk (async, no VGPRs)
    compute(cur);                              // 16 WMMAs on current chunk
    if (kc + 1 < KCH) {
      wait_asynccnt0();                        // next-chunk data landed in LDS
      __syncthreads();
      cur ^= 1;
    }
  }

  // epilogue: D VGPR r -> row r + 8h, col = lane&15
#pragma unroll
  for (int mt = 0; mt < 4; ++mt) {
    int rowbase = bm0 + wm * 64 + mt * 16 + 8 * h;
#pragma unroll
    for (int nt = 0; nt < 2; ++nt) {
      int col = on0 + wn * 32 + nt * 16 + nl;
      float bv = bias[col];
      v8f c = acc[mt][nt];
#pragma unroll
      for (int r = 0; r < 8; ++r)
        out[((size_t)(rowbase + r) << 12) + col] = c[r] + bv;
    }
  }
}

// ---------------------------------------------------------------------------
// Host launcher
// ---------------------------------------------------------------------------
extern "C" void kernel_launch(void* const* d_in, const int* in_sizes, int n_in,
                              void* d_out, int out_size, void* d_ws, size_t ws_size,
                              hipStream_t stream) {
  const float* input   = (const float*)d_in[0];
  const int*   qweight = (const int*)  d_in[1];
  const int*   scale_q = (const int*)  d_in[2];
  const float* sqs     = (const float*)d_in[3];
  const float* sqz     = (const float*)d_in[4];
  const int*   zero_q  = (const int*)  d_in[5];
  const float* zqs     = (const float*)d_in[6];
  const float* zqz     = (const float*)d_in[7];
  const int*   orow    = (const int*)  d_in[8];
  const int*   ocol    = (const int*)  d_in[9];
  const float* oval    = (const float*)d_in[10];
  const float* bias    = (const float*)d_in[11];
  const int*   perm    = (const int*)  d_in[12];
  float*       out     = (float*)d_out;

  // workspace: x_bf (32 MB) then W_bf (32 MB)
  unsigned short* xbf = (unsigned short*)d_ws;
  unsigned short* wbf = xbf + (size_t)Bt * Id;

  // 1) permute + f32->bf16 convert of x
  k_permute_x<<<(Bt * (size_t)Id) / 256, 256, 0, stream>>>(input, perm, xbf);

  // 2) dequantize 4-bit weights -> bf16 W
  k_dequant_w<<<((size_t)Od * (Id / 2)) / 256, 256, 0, stream>>>(
      qweight, scale_q, sqs, sqz, zero_q, zqs, zqz, wbf);

  // 3) zero outlier slots, 4) fold outlier values into W
  k_zero_outliers<<<(NNZ + 255) / 256, 256, 0, stream>>>(wbf, orow, ocol);
  k_add_outliers<<<(NNZ + 255) / 256, 256, 0, stream>>>(wbf, orow, ocol, oval);

  // 5) WMMA GEMM with fused bias
  dim3 grid(Od / BN, Bt / BM);
  k_gemm_bf16<<<grid, 256, 0, stream>>>(xbf, wbf, bias, out);
}
